// CausalSelfAttention_22789096473012
// MI455X (gfx1250) — matrix operand order repaired and checked by reading, not compiled, not run on hardware
//
#include <hip/hip_runtime.h>
#include <math.h>

typedef __attribute__((ext_vector_type(16))) _Float16 v16h;
typedef __attribute__((ext_vector_type(8)))  float    v8f;
typedef __attribute__((ext_vector_type(4)))  unsigned u32x4;
typedef __attribute__((ext_vector_type(4)))  int      v4i;

typedef __attribute__((address_space(1))) v4i* gptr_v4i;
typedef __attribute__((address_space(3))) v4i* lptr_v4i;

union Frag16 { v16h h; u32x4 q[2]; };

#define T_SEQ  2048
#define DIMSZ  1024
#define NHEAD  8
#define HD     128
#define QKVN   3072
#define ATTN_SCALE 0.12f
#define RMS_EPS 1.192092896e-7f

// ---- CDNA5 async global->LDS path (probe via __has_builtin) -----------------
#if defined(__has_builtin)
#if __has_builtin(__builtin_amdgcn_global_load_async_to_lds_b128) && \
    __has_builtin(__builtin_amdgcn_s_wait_asynccnt)
#define HAVE_ASYNC 1
#endif
#endif
#ifndef HAVE_ASYNC
#define HAVE_ASYNC 0
#endif

__device__ __forceinline__ void async_cp16(_Float16* dst, const _Float16* src) {
#if HAVE_ASYNC
    __builtin_amdgcn_global_load_async_to_lds_b128(
        (gptr_v4i)(void*)src, (lptr_v4i)(void*)dst, 0, 0);
#else
    *(u32x4*)dst = *(const u32x4*)src;
#endif
}

__device__ __forceinline__ void wait_async() {
#if HAVE_ASYNC
    __builtin_amdgcn_s_wait_asynccnt(0);
#endif
}

// ---------------------------------------------------------------------------
// fp32 -> f16 convert (grid-stride)
// ---------------------------------------------------------------------------
__global__ void cvt_f32_f16(const float* __restrict__ in,
                            _Float16* __restrict__ out, int n) {
    int i = blockIdx.x * blockDim.x + threadIdx.x;
    int stride = gridDim.x * blockDim.x;
    for (; i < n; i += stride) out[i] = (_Float16)in[i];
}

// ---------------------------------------------------------------------------
// WMMA GEMM: C[M,N] = A[M,K] * B[N,K]^T  (both A,B row-major over K, f16)
// 128x128 tile per block, 8 waves; K sliced in 32-steps through a
// double-buffered LDS pair staged with async global->LDS copies.
// ---------------------------------------------------------------------------
__global__ __launch_bounds__(256)
void gemm_f16_wmma(const _Float16* __restrict__ A,
                   const _Float16* __restrict__ B,
                   float* __restrict__ C, int M, int N, int K) {
    __shared__ _Float16 ldsA[2][128 * 32];
    __shared__ _Float16 ldsB[2][128 * 32];

    const int tid  = threadIdx.x;
    const int w    = tid >> 5;
    const int lane = tid & 31;
    const int half = lane >> 4;
    const int nl   = lane & 15;
    const int m0   = blockIdx.y * 128;
    const int n0   = blockIdx.x * 128;

    // staging map: chunk i in [0,512): row = i>>2, col = (i&3)*8 f16 (16B)
    const int r0 = tid >> 2,         c0 = (tid & 3) * 8;
    const int r1 = (tid + 256) >> 2, c1 = ((tid + 256) & 3) * 8;

    v8f acc[8];
    const v8f vzero = {0.f,0.f,0.f,0.f,0.f,0.f,0.f,0.f};
#pragma unroll
    for (int i = 0; i < 8; ++i) acc[i] = vzero;

    // prologue: stage slab 0 into buffer 0
    async_cp16(&ldsA[0][r0 * 32 + c0], &A[(size_t)(m0 + r0) * K + c0]);
    async_cp16(&ldsA[0][r1 * 32 + c1], &A[(size_t)(m0 + r1) * K + c1]);
    async_cp16(&ldsB[0][r0 * 32 + c0], &B[(size_t)(n0 + r0) * K + c0]);
    async_cp16(&ldsB[0][r1 * 32 + c1], &B[(size_t)(n0 + r1) * K + c1]);
    wait_async();
    __syncthreads();

    const int nk = K >> 5;
    for (int t = 0; t < nk; ++t) {
        const int buf = t & 1;
        // stage next slab into the other buffer while we compute
        if (t + 1 < nk) {
            const int kk = (t + 1) * 32;
            async_cp16(&ldsA[buf ^ 1][r0 * 32 + c0], &A[(size_t)(m0 + r0) * K + kk + c0]);
            async_cp16(&ldsA[buf ^ 1][r1 * 32 + c1], &A[(size_t)(m0 + r1) * K + kk + c1]);
            async_cp16(&ldsB[buf ^ 1][r0 * 32 + c0], &B[(size_t)(n0 + r0) * K + kk + c0]);
            async_cp16(&ldsB[buf ^ 1][r1 * 32 + c1], &B[(size_t)(n0 + r1) * K + kk + c1]);
        }

        // A fragment (wave's 16 rows x 32 K)
        Frag16 a;
        {
            const _Float16* p = &ldsA[buf][(w * 16 + nl) * 32 + 8 * half];
            a.q[0] = *(const u32x4*)p;
            a.q[1] = *(const u32x4*)(p + 16);
        }
        // preload all 8 B fragments, then chain the WMMAs back-to-back
        Frag16 b[8];
#pragma unroll
        for (int nt = 0; nt < 8; ++nt) {
            const _Float16* p = &ldsB[buf][(nt * 16 + nl) * 32 + 8 * half];
            b[nt].q[0] = *(const u32x4*)p;
            b[nt].q[1] = *(const u32x4*)(p + 16);
        }
#pragma unroll
        for (int nt = 0; nt < 8; ++nt)
            acc[nt] = __builtin_amdgcn_wmma_f32_16x16x32_f16(
                false, a.h, false, b[nt].h, (short)0, acc[nt], false, false);

        wait_async();       // my async writes to buf^1 have landed
        __syncthreads();    // everyone done reading buf / writing buf^1
    }

#pragma unroll
    for (int nt = 0; nt < 8; ++nt)
#pragma unroll
        for (int r = 0; r < 8; ++r)
            C[(size_t)(m0 + w * 16 + r + 8 * half) * N + n0 + nt * 16 + nl] =
                acc[nt][r];
}

// ---------------------------------------------------------------------------
// Post-QKV: RMS-norm q,k; RoPE; v = l0*v + l1*ve.
// One block per (t, head), 128 threads.
// Outputs: qh,kh as [h][T][128] f16 ; v transposed as [h][128][T] f16.
// ---------------------------------------------------------------------------
__global__ __launch_bounds__(128)
void post_qkv(const float* __restrict__ qkv, const float* __restrict__ ve,
              const float* __restrict__ lambdas,
              _Float16* __restrict__ qh, _Float16* __restrict__ kh,
              _Float16* __restrict__ vt) {
    const int t = blockIdx.x, h = blockIdx.y, d = threadIdx.x;
    __shared__ float qs[128], ks[128], red[128];

    const size_t base = (size_t)t * QKVN + (size_t)h * HD + d;
    float qv = qkv[base];
    float kv = qkv[base + 1024];
    float vv = qkv[base + 2048];

    red[d] = qv * qv;  __syncthreads();
    for (int s = 64; s > 0; s >>= 1) {
        if (d < s) red[d] += red[d + s];
        __syncthreads();
    }
    float qn = qv * rsqrtf(red[0] * (1.0f / 128.0f) + RMS_EPS);
    __syncthreads();
    red[d] = kv * kv;  __syncthreads();
    for (int s = 64; s > 0; s >>= 1) {
        if (d < s) red[d] += red[d + s];
        __syncthreads();
    }
    float kn = kv * rsqrtf(red[0] * (1.0f / 128.0f) + RMS_EPS);
    __syncthreads();

    qs[d] = qn;  ks[d] = kn;
    __syncthreads();

    const int i = d & 63;
    float c, s;
    if (i < 32) {
        float f  = powf(1024.0f, -(float)i * (1.0f / 31.0f));
        float th = (float)t * f;
        c = cosf(th);  s = sinf(th);
    } else { c = 1.0f; s = 0.0f; }

    float x1q = qs[i], x2q = qs[i + 64];
    float x1k = ks[i], x2k = ks[i + 64];
    float oq = (d < 64) ? (x1q * c + x2q * s) : (-x1q * s + x2q * c);
    float ok = (d < 64) ? (x1k * c + x2k * s) : (-x1k * s + x2k * c);

    const size_t oidx = ((size_t)h * T_SEQ + t) * HD + d;
    qh[oidx] = (_Float16)oq;
    kh[oidx] = (_Float16)ok;

    float vb = lambdas[0] * vv + lambdas[1] * ve[(size_t)t * DIMSZ + h * HD + d];
    vt[((size_t)h * HD + d) * T_SEQ + t] = (_Float16)vb;
}

// ---------------------------------------------------------------------------
// Flash attention (causal, online softmax).
// grid = (T/128, H), 256 threads; each wave owns 16 query rows.
// ---------------------------------------------------------------------------
__global__ __launch_bounds__(256)
void flash_attn(const _Float16* __restrict__ qh, const _Float16* __restrict__ kh,
                const _Float16* __restrict__ vt, _Float16* __restrict__ yh) {
    __shared__ _Float16 ldsP[8 * 16 * 32];   // per-wave 16x32 P staging

    const int tid  = threadIdx.x;
    const int w    = tid >> 5;
    const int lane = tid & 31;
    const int half = lane >> 4;
    const int nl   = lane & 15;
    const int h     = blockIdx.y;
    const int qBase = blockIdx.x * 128 + w * 16;

    const _Float16* qp = qh + (size_t)h * T_SEQ * HD;
    const _Float16* kp = kh + (size_t)h * T_SEQ * HD;
    const _Float16* vp = vt + (size_t)h * HD * T_SEQ;

    Frag16 aQ[4];
#pragma unroll
    for (int kc = 0; kc < 4; ++kc) {
        const _Float16* p = qp + (size_t)(qBase + nl) * HD + kc * 32 + 8 * half;
        aQ[kc].q[0] = *(const u32x4*)p;
        aQ[kc].q[1] = *(const u32x4*)(p + 16);
    }

    const v8f vzero = {0.f,0.f,0.f,0.f,0.f,0.f,0.f,0.f};
    v8f o[8];
    float mr[8], lsum[8];
#pragma unroll
    for (int i = 0; i < 8; ++i) { o[i] = vzero; mr[i] = -3.0e38f; lsum[i] = 0.0f; }

    _Float16* myP = &ldsP[w * 16 * 32];

    for (int s0 = 0; s0 < qBase + 16; s0 += 32) {
        // ---- S = Q K^T : preload K fragments, then chain WMMAs
        v8f sf[2];  sf[0] = vzero;  sf[1] = vzero;
#pragma unroll
        for (int st = 0; st < 2; ++st) {
            Frag16 bk[4];
#pragma unroll
            for (int kc = 0; kc < 4; ++kc) {
                const _Float16* p =
                    kp + (size_t)(s0 + st * 16 + nl) * HD + kc * 32 + 8 * half;
                bk[kc].q[0] = *(const u32x4*)p;
                bk[kc].q[1] = *(const u32x4*)(p + 16);
            }
#pragma unroll
            for (int kc = 0; kc < 4; ++kc)
                sf[st] = __builtin_amdgcn_wmma_f32_16x16x32_f16(
                    false, aQ[kc].h, false, bk[kc].h, (short)0, sf[st], false, false);
        }

        const bool needMask = (s0 + 31 > qBase);
        float pbuf[2][8];
        float alpha[8];
#pragma unroll
        for (int r = 0; r < 8; ++r) {
            const int srow = qBase + r + 8 * half;
            float a0 = sf[0][r] * ATTN_SCALE;
            float a1 = sf[1][r] * ATTN_SCALE;
            if (needMask) {
                if (s0 + nl      > srow) a0 = -3.0e38f;
                if (s0 + 16 + nl > srow) a1 = -3.0e38f;
            }
            float m = fmaxf(a0, a1);
            m = fmaxf(m, __shfl_xor(m, 1, 32));
            m = fmaxf(m, __shfl_xor(m, 2, 32));
            m = fmaxf(m, __shfl_xor(m, 4, 32));
            m = fmaxf(m, __shfl_xor(m, 8, 32));
            float mn = fmaxf(mr[r], m);
            float al = __expf(mr[r] - mn);
            mr[r] = mn;
            float p0 = __expf(a0 - mn);
            float p1 = __expf(a1 - mn);
            pbuf[0][r] = p0;  pbuf[1][r] = p1;
            float ls = p0 + p1;
            ls += __shfl_xor(ls, 1, 32);
            ls += __shfl_xor(ls, 2, 32);
            ls += __shfl_xor(ls, 4, 32);
            ls += __shfl_xor(ls, 8, 32);
            lsum[r] = lsum[r] * al + ls;
            alpha[r] = al;
        }
#pragma unroll
        for (int nt = 0; nt < 8; ++nt)
#pragma unroll
            for (int r = 0; r < 8; ++r)
                o[nt][r] *= alpha[r];

        // ---- C-layout -> A-layout transpose of P through wave-private LDS
#pragma unroll
        for (int st = 0; st < 2; ++st)
#pragma unroll
            for (int r = 0; r < 8; ++r)
                myP[(r + 8 * half) * 32 + st * 16 + nl] = (_Float16)pbuf[st][r];
        __asm__ volatile("s_wait_dscnt 0" ::: "memory");

        Frag16 pf;
        {
            const _Float16* p = &myP[nl * 32 + 8 * half];
            pf.q[0] = *(const u32x4*)p;
            pf.q[1] = *(const u32x4*)(p + 16);
        }

        // ---- O += P * V : preload V fragments, then chain WMMAs
        Frag16 bv[8];
#pragma unroll
        for (int nt = 0; nt < 8; ++nt) {
            const _Float16* p = vp + (size_t)(nt * 16 + nl) * T_SEQ + s0 + 8 * half;
            bv[nt].q[0] = *(const u32x4*)p;
            bv[nt].q[1] = *(const u32x4*)(p + 16);
        }
#pragma unroll
        for (int nt = 0; nt < 8; ++nt)
            o[nt] = __builtin_amdgcn_wmma_f32_16x16x32_f16(
                false, pf.h, false, bv[nt].h, (short)0, o[nt], false, false);
    }

#pragma unroll
    for (int r = 0; r < 8; ++r) lsum[r] = 1.0f / lsum[r];
#pragma unroll
    for (int nt = 0; nt < 8; ++nt)
#pragma unroll
        for (int r = 0; r < 8; ++r)
            yh[(size_t)(qBase + r + 8 * half) * DIMSZ + h * HD + nt * 16 + nl] =
                (_Float16)(o[nt][r] * lsum[r]);
}

// ---------------------------------------------------------------------------
extern "C" void kernel_launch(void* const* d_in, const int* in_sizes, int n_in,
                              void* d_out, int out_size, void* d_ws, size_t ws_size,
                              hipStream_t stream) {
    const float* x        = (const float*)d_in[0];  // [1,2048,1024]
    const float* ve       = (const float*)d_in[1];  // [1,2048,1024]
    const float* lambdas  = (const float*)d_in[2];  // [2]
    const float* qkv_w    = (const float*)d_in[3];  // [3,1024,1024]
    const float* c_proj_w = (const float*)d_in[4];  // [1024,1024]
    float* out = (float*)d_out;                     // [1,2048,1024]

    char* ws = (char*)d_ws;
    size_t off = 0;
    _Float16* xh     = (_Float16*)(ws + off); off += (size_t)T_SEQ * DIMSZ * 2;
    _Float16* wqkvh  = (_Float16*)(ws + off); off += (size_t)3 * DIMSZ * DIMSZ * 2;
    _Float16* wprojh = (_Float16*)(ws + off); off += (size_t)DIMSZ * DIMSZ * 2;
    float*    qkv    = (float*)   (ws + off); off += (size_t)T_SEQ * QKVN * 4;
    _Float16* qh     = (_Float16*)(ws + off); off += (size_t)NHEAD * T_SEQ * HD * 2;
    _Float16* kh     = (_Float16*)(ws + off); off += (size_t)NHEAD * T_SEQ * HD * 2;
    _Float16* vt     = (_Float16*)(ws + off); off += (size_t)NHEAD * HD * T_SEQ * 2;
    _Float16* yh     = (_Float16*)(ws + off); off += (size_t)T_SEQ * DIMSZ * 2;

    cvt_f32_f16<<<2048, 256, 0, stream>>>(x, xh, T_SEQ * DIMSZ);
    cvt_f32_f16<<<2048, 256, 0, stream>>>(qkv_w, wqkvh, 3 * DIMSZ * DIMSZ);
    cvt_f32_f16<<<2048, 256, 0, stream>>>(c_proj_w, wprojh, DIMSZ * DIMSZ);

    gemm_f16_wmma<<<dim3(QKVN / 128, T_SEQ / 128), 256, 0, stream>>>(
        xh, wqkvh, qkv, T_SEQ, QKVN, DIMSZ);

    post_qkv<<<dim3(T_SEQ, NHEAD), 128, 0, stream>>>(qkv, ve, lambdas, qh, kh, vt);

    flash_attn<<<dim3(T_SEQ / 128, NHEAD), 256, 0, stream>>>(qh, kh, vt, yh);

    gemm_f16_wmma<<<dim3(DIMSZ / 128, T_SEQ / 128), 256, 0, stream>>>(
        yh, wprojh, out, T_SEQ, DIMSZ, DIMSZ);
}
